// Pointnet2MSG_46454366274096
// MI455X (gfx1250) — compile-verified, matrix operand order checked by reading, not run on hardware
//
#include <hip/hip_runtime.h>
#include <hip/hip_bf16.h>

// ---------------------------------------------------------------------------
// CDNA5 WMMA types (wave32): D(16x16 f32) = A(16x32 bf16) x B(32x16 bf16) + C
// ---------------------------------------------------------------------------
typedef __attribute__((ext_vector_type(16))) __bf16         v16bf;
typedef __attribute__((ext_vector_type(16))) unsigned short v16s;
typedef __attribute__((ext_vector_type(8)))  unsigned short v8s;
typedef __attribute__((ext_vector_type(8)))  float          v8f;

__device__ __forceinline__ unsigned short f2bf(float x) {
  unsigned u = __builtin_bit_cast(unsigned, x);
  unsigned r = u + 0x7fffu + ((u >> 16) & 1u);   // round-to-nearest-even
  return (unsigned short)(r >> 16);
}

// ---------------------------------------------------------------------------
// GEMM: C[M,N] = A[M,Kpad](bf16) * Wt[Npad32,Kpad](bf16, pre-transposed)
//       (+bias)(+relu).  M % 16 == 0, Kpad % 32 == 0, Wt zero-padded.
// One 16x32 output block per wave (two 16x16 WMMA tiles sharing the A frag).
// Inner loop: 6x global_load_b128 + 2x v_wmma, no guards, no conversion.
// ---------------------------------------------------------------------------
__global__ void gemm_bf16_wmma(const unsigned short* __restrict__ A,
                               const unsigned short* __restrict__ Wt,
                               const float* __restrict__ bias,
                               float* __restrict__ C,
                               int M, int N, int Kpad, int do_relu) {
  const int lane = threadIdx.x & 31;
  const int l15  = lane & 15;
  const int ksel = (lane >> 4) << 3;          // 0 or 8: K-half held by this lane
  const long long wave  = (long long)blockIdx.x * (blockDim.x >> 5) + (threadIdx.x >> 5);
  const long long nwave = (long long)gridDim.x * (blockDim.x >> 5);
  const int Mt  = M >> 4;
  const int Nt2 = (N + 31) >> 5;
  const long long tiles = (long long)Mt * Nt2;

  for (long long t = wave; t < tiles; t += nwave) {
    const int m0 = (int)(t % Mt) << 4;
    const int n0 = (int)(t / Mt) << 5;
    const unsigned short* arow  = A  + (size_t)(m0 + l15) * Kpad + ksel;
    const unsigned short* brow0 = Wt + (size_t)(n0 + l15) * Kpad + ksel;
    const unsigned short* brow1 = brow0 + (size_t)16 * Kpad;
    v8f acc0 = {}, acc1 = {};

    for (int k0 = 0; k0 < Kpad; k0 += 32) {
      const v8s alo = *(const v8s*)(arow + k0);
      const v8s ahi = *(const v8s*)(arow + k0 + 16);
      const v8s b0l = *(const v8s*)(brow0 + k0);
      const v8s b0h = *(const v8s*)(brow0 + k0 + 16);
      const v8s b1l = *(const v8s*)(brow1 + k0);
      const v8s b1h = *(const v8s*)(brow1 + k0 + 16);
      const v16s au  = __builtin_shufflevector(alo, ahi, 0,1,2,3,4,5,6,7,8,9,10,11,12,13,14,15);
      const v16s bu0 = __builtin_shufflevector(b0l, b0h, 0,1,2,3,4,5,6,7,8,9,10,11,12,13,14,15);
      const v16s bu1 = __builtin_shufflevector(b1l, b1h, 0,1,2,3,4,5,6,7,8,9,10,11,12,13,14,15);
      acc0 = __builtin_amdgcn_wmma_f32_16x16x32_bf16(
          false, __builtin_bit_cast(v16bf, au), false, __builtin_bit_cast(v16bf, bu0),
          (short)0, acc0, false, false);
      acc1 = __builtin_amdgcn_wmma_f32_16x16x32_bf16(
          false, __builtin_bit_cast(v16bf, au), false, __builtin_bit_cast(v16bf, bu1),
          (short)0, acc1, false, false);
    }

    // D layout: VGPR r -> row m0 + r + (lane>=16 ? 8 : 0); col = n-base + (lane&15)
    const int rbase = m0 + ((lane >> 4) << 3);
    {
      const int ccol = n0 + l15;
      if (ccol < N) {
        const float bv = bias ? bias[ccol] : 0.f;
#pragma unroll
        for (int r = 0; r < 8; ++r) {
          float v = acc0[r] + bv;
          if (do_relu) v = v > 0.f ? v : 0.f;
          C[(size_t)(rbase + r) * N + ccol] = v;
        }
      }
    }
    {
      const int ccol = n0 + 16 + l15;
      if (ccol < N) {
        const float bv = bias ? bias[ccol] : 0.f;
#pragma unroll
        for (int r = 0; r < 8; ++r) {
          float v = acc1[r] + bv;
          if (do_relu) v = v > 0.f ? v : 0.f;
          C[(size_t)(rbase + r) * N + ccol] = v;
        }
      }
    }
  }
}

// W f32 [K,N] -> Wt bf16 [Npad32, Kpad], transposed, zero padded
__global__ void wt_convert_kernel(const float* __restrict__ W, int K, int N,
                                  int Kpad, int Npad, unsigned short* __restrict__ Wt) {
  const long long total  = (long long)Npad * Kpad;
  const long long stride = (long long)gridDim.x * blockDim.x;
  for (long long g = (long long)blockIdx.x * blockDim.x + threadIdx.x; g < total; g += stride) {
    const int n = (int)(g / Kpad);
    const int k = (int)(g % Kpad);
    Wt[g] = (n < N && k < K) ? f2bf(W[(size_t)k * N + n]) : (unsigned short)0;
  }
}

// ---------------------------------------------------------------------------
// BatchNorm stats: one block per channel
// ---------------------------------------------------------------------------
__global__ void bn_stats_kernel(const float* __restrict__ Y, int M, int N,
                                float* __restrict__ mean, float* __restrict__ rstd) {
  const int c = blockIdx.x;
  __shared__ float ss[256], sq[256];
  float s = 0.f, q = 0.f;
  for (int m = threadIdx.x; m < M; m += 256) {
    float v = Y[(size_t)m * N + c];
    s += v; q += v * v;
  }
  ss[threadIdx.x] = s; sq[threadIdx.x] = q;
  __syncthreads();
  for (int st = 128; st > 0; st >>= 1) {
    if (threadIdx.x < st) { ss[threadIdx.x] += ss[threadIdx.x + st]; sq[threadIdx.x] += sq[threadIdx.x + st]; }
    __syncthreads();
  }
  if (threadIdx.x == 0) {
    float mu  = ss[0] / (float)M;
    float var = sq[0] / (float)M - mu * mu;
    if (var < 0.f) var = 0.f;
    mean[c] = mu;
    rstd[c] = rsqrtf(var + 1e-5f);
  }
}

// BN + ReLU, dual output: optional f32 [M,N] and optional padded bf16 [M,Kpad]
__global__ void bn_relu_dual_kernel(const float* __restrict__ Y, const float* __restrict__ mean,
                                    const float* __restrict__ rstd, const float* __restrict__ gam,
                                    const float* __restrict__ bet, float* __restrict__ dstF,
                                    unsigned short* __restrict__ dstB,
                                    long long Mrows, int N, int Kpad) {
  const long long total  = Mrows * Kpad;
  const long long stride = (long long)gridDim.x * blockDim.x;
  for (long long g = (long long)blockIdx.x * blockDim.x + threadIdx.x; g < total; g += stride) {
    const long long row = g / Kpad;
    const int c = (int)(g % Kpad);
    if (c < N) {
      float v = (Y[row * N + c] - mean[c]) * rstd[c] * gam[c] + bet[c];
      v = v > 0.f ? v : 0.f;
      if (dstF) dstF[row * N + c] = v;
      if (dstB) dstB[g] = f2bf(v);
    } else if (dstB) {
      dstB[g] = 0;
    }
  }
}

// ---------------------------------------------------------------------------
// Pointcloud split [B,N,9] -> xyz [B,N,3] + feats [B,N,6]
// ---------------------------------------------------------------------------
__global__ void split_pc_kernel(const float* __restrict__ pc, int total,
                                float* __restrict__ xyz, float* __restrict__ feats) {
  const int g = blockIdx.x * blockDim.x + threadIdx.x;
  if (g >= total) return;
  const float* p = pc + (size_t)g * 9;
  xyz[(size_t)g * 3 + 0] = p[0];
  xyz[(size_t)g * 3 + 1] = p[1];
  xyz[(size_t)g * 3 + 2] = p[2];
#pragma unroll
  for (int c = 0; c < 6; ++c) feats[(size_t)g * 6 + c] = p[3 + c];
}

// ---------------------------------------------------------------------------
// Farthest point sampling: one block (256 thr) per batch
// ---------------------------------------------------------------------------
__global__ void fps_kernel(const float* __restrict__ xyz, int Nn, int npoint,
                           int* __restrict__ oidx, float* __restrict__ dist) {
  const int b = blockIdx.x, t = threadIdx.x;
  const float* X = xyz + (size_t)b * Nn * 3;
  float* D = dist + (size_t)b * Nn;
  int*   O = oidx + (size_t)b * npoint;
  __shared__ float sv[256];
  __shared__ int   si[256];
  __shared__ float sc[3];
  for (int n = t; n < Nn; n += 256) D[n] = 1e10f;
  if (t == 0) { sc[0] = X[0]; sc[1] = X[1]; sc[2] = X[2]; }
  __syncthreads();
  int far = 0;
  for (int it = 0; it < npoint; ++it) {
    if (t == 0) O[it] = far;
    const float cx = sc[0], cy = sc[1], cz = sc[2];
    float best = -1.f; int bi = 0;
    for (int n = t; n < Nn; n += 256) {
      float dx = X[3 * n] - cx, dy = X[3 * n + 1] - cy, dz = X[3 * n + 2] - cz;
      float d = dx * dx + dy * dy + dz * dz;
      float dd = D[n];
      if (d < dd) { dd = d; D[n] = dd; }
      if (dd > best) { best = dd; bi = n; }
    }
    sv[t] = best; si[t] = bi;
    __syncthreads();
    for (int s2 = 128; s2 > 0; s2 >>= 1) {
      if (t < s2 && sv[t + s2] > sv[t]) { sv[t] = sv[t + s2]; si[t] = si[t + s2]; }
      __syncthreads();
    }
    far = si[0];
    if (t == 0) { sc[0] = X[3 * far]; sc[1] = X[3 * far + 1]; sc[2] = X[3 * far + 2]; }
    __syncthreads();
  }
}

__global__ void gather_xyz_kernel(const float* __restrict__ xyz, const int* __restrict__ fidx,
                                  int total, int Nn, int np, float* __restrict__ out) {
  const int g = blockIdx.x * blockDim.x + threadIdx.x;
  if (g >= total) return;
  const int b = g / np;
  const int id = fidx[g];
  const float* s = xyz + ((size_t)b * Nn + id) * 3;
  out[(size_t)g * 3 + 0] = s[0];
  out[(size_t)g * 3 + 1] = s[1];
  out[(size_t)g * 3 + 2] = s[2];
}

// ---------------------------------------------------------------------------
// Ball query: first ns indices within radius (ascending), pad with first hit
// ---------------------------------------------------------------------------
__global__ void ball_query_kernel(const float* __restrict__ xyz, const float* __restrict__ nxyz,
                                  int Bn, int Nn, int np, int ns, float r2,
                                  int* __restrict__ idx) {
  const int g = blockIdx.x * blockDim.x + threadIdx.x;
  if (g >= Bn * np) return;
  const int b = g / np;
  const float* X = xyz + (size_t)b * Nn * 3;
  const float* c = nxyz + (size_t)g * 3;
  int* o = idx + (size_t)g * ns;
  int cnt = 0, first = -1;
  for (int n = 0; n < Nn && cnt < ns; ++n) {
    float dx = X[3 * n] - c[0], dy = X[3 * n + 1] - c[1], dz = X[3 * n + 2] - c[2];
    if (dx * dx + dy * dy + dz * dz < r2) {
      if (first < 0) first = n;
      o[cnt++] = n;
    }
  }
  if (first < 0) first = 0;
  for (; cnt < ns; ++cnt) o[cnt] = first;
}

// grouped input, written directly as padded bf16 GEMM A-matrix [rows, Kpad]
__global__ void group_bf16_kernel(const float* __restrict__ xyz, const float* __restrict__ feats,
                                  const float* __restrict__ nxyz, const int* __restrict__ idx,
                                  long long rows, int Nn, int np, int ns, int C, int Kpad,
                                  unsigned short* __restrict__ out) {
  const long long total  = rows * Kpad;
  const long long stride = (long long)gridDim.x * blockDim.x;
  for (long long g = (long long)blockIdx.x * blockDim.x + threadIdx.x; g < total; g += stride) {
    const long long row = g / Kpad;
    const int c = (int)(g % Kpad);
    unsigned short v = 0;
    if (c < 3 + C) {
      const int p = (int)((row / ns) % np);
      const int b = (int)(row / ((long long)ns * np));
      const int id = idx[row];
      if (c < 3) {
        const float* X  = xyz + ((size_t)b * Nn + id) * 3;
        const float* cc = nxyz + ((size_t)b * np + p) * 3;
        v = f2bf(X[c] - cc[c]);
      } else {
        v = f2bf(feats[((size_t)b * Nn + id) * C + (c - 3)]);
      }
    }
    out[g] = v;
  }
}

// max over nsample of f32 activations, written into channel slice of level feats
__global__ void maxpool_kernel(const float* __restrict__ X, long long centers, int ns, int C,
                               float* __restrict__ out, int ldo, int coff) {
  const long long total  = centers * C;
  const long long stride = (long long)gridDim.x * blockDim.x;
  for (long long g = (long long)blockIdx.x * blockDim.x + threadIdx.x; g < total; g += stride) {
    const long long r = g / C;
    const int c = (int)(g % C);
    const float* p = X + (size_t)(r * ns) * C + c;
    float m = -1e30f;
    for (int s = 0; s < ns; ++s) { float v = p[(size_t)s * C]; if (v > m) m = v; }
    out[(size_t)r * ldo + coff + c] = m;
  }
}

// ---------------------------------------------------------------------------
// 3-NN + inverse-distance weights (FP module)
// ---------------------------------------------------------------------------
__global__ void knn3_kernel(const float* __restrict__ U, const float* __restrict__ Kx,
                            int Bn, int Nu, int Nk, int* __restrict__ oi, float* __restrict__ ow) {
  const long long g = (long long)blockIdx.x * blockDim.x + threadIdx.x;
  if (g >= (long long)Bn * Nu) return;
  const int b = (int)(g / Nu);
  const float* u = U + (size_t)g * 3;
  const float* P = Kx + (size_t)b * Nk * 3;
  float d0 = 1e30f, d1 = 1e30f, d2 = 1e30f;
  int i0 = 0, i1 = 0, i2 = 0;
  for (int n = 0; n < Nk; ++n) {
    float dx = P[3 * n] - u[0], dy = P[3 * n + 1] - u[1], dz = P[3 * n + 2] - u[2];
    float d = dx * dx + dy * dy + dz * dz;
    if (d < d0)      { d2 = d1; i2 = i1; d1 = d0; i1 = i0; d0 = d; i0 = n; }
    else if (d < d1) { d2 = d1; i2 = i1; d1 = d; i1 = n; }
    else if (d < d2) { d2 = d; i2 = n; }
  }
  float w0 = 1.f / (d0 + 1e-8f), w1 = 1.f / (d1 + 1e-8f), w2 = 1.f / (d2 + 1e-8f);
  const float s = w0 + w1 + w2;
  w0 /= s; w1 /= s; w2 /= s;
  oi[g * 3 + 0] = i0; oi[g * 3 + 1] = i1; oi[g * 3 + 2] = i2;
  ow[g * 3 + 0] = w0; ow[g * 3 + 1] = w1; ow[g * 3 + 2] = w2;
}

// [interp | skip] concat written directly as padded bf16 GEMM A-matrix
__global__ void interp_concat_bf16_kernel(const float* __restrict__ KF, const int* __restrict__ ki,
                                          const float* __restrict__ kw, const float* __restrict__ SF,
                                          int Bn, int Nu, int Nk, int Ck, int Cs, int Kpad,
                                          unsigned short* __restrict__ out) {
  const long long total  = (long long)Bn * Nu * Kpad;
  const long long stride = (long long)gridDim.x * blockDim.x;
  for (long long g = (long long)blockIdx.x * blockDim.x + threadIdx.x; g < total; g += stride) {
    const long long row = g / Kpad;
    const int c = (int)(g % Kpad);
    unsigned short v = 0;
    if (c < Ck) {
      const int b = (int)(row / Nu);
      const int*   ii = ki + row * 3;
      const float* ww = kw + row * 3;
      const float* F  = KF + (size_t)b * Nk * Ck;
      v = f2bf(ww[0] * F[(size_t)ii[0] * Ck + c] +
               ww[1] * F[(size_t)ii[1] * Ck + c] +
               ww[2] * F[(size_t)ii[2] * Ck + c]);
    } else if (c < Ck + Cs) {
      v = f2bf(SF[(size_t)row * Cs + (c - Ck)]);
    }
    out[g] = v;
  }
}

// final 128 -> 1 head
__global__ void fc2_kernel(const float* __restrict__ h, const float* __restrict__ W,
                           const float* __restrict__ b, float* __restrict__ pred, int rows) {
  const int g = blockIdx.x * blockDim.x + threadIdx.x;
  if (g >= rows) return;
  float s = b[0];
  for (int c = 0; c < 128; ++c) s += h[(size_t)g * 128 + c] * W[c];
  pred[g] = s;
}

// ---------------------------------------------------------------------------
// Host orchestration
// ---------------------------------------------------------------------------
namespace {
struct Layer { const float *W, *b, *g, *beta; };

inline unsigned cdivu(long long n, int b) {
  long long g = (n + b - 1) / b;
  if (g < 1) g = 1;
  return (unsigned)g;
}
inline unsigned capped(long long n, int b, long long cap) {
  long long g = (n + b - 1) / b;
  if (g < 1) g = 1;
  if (g > cap) g = cap;
  return (unsigned)g;
}
inline int ceil32(int x) { return (x + 31) & ~31; }
}  // namespace

extern "C" void kernel_launch(void* const* d_in, const int* in_sizes, int n_in,
                              void* d_out, int out_size, void* d_ws, size_t ws_size,
                              hipStream_t stream) {
  (void)in_sizes; (void)n_in; (void)out_size; (void)ws_size;

  const int Bn = 2, Nn = 16384;
  static const int   SA_NP[4]   = {4096, 1024, 256, 64};
  static const float SA_R[4][2] = {{0.1f, 0.5f}, {0.5f, 1.0f}, {1.0f, 2.0f}, {2.0f, 4.0f}};
  static const int   SA_NS[2]   = {16, 32};
  static const int   SA_DIMS[4][2][4] = {
      {{9, 16, 16, 32},      {9, 32, 32, 64}},
      {{99, 64, 64, 128},    {99, 64, 96, 128}},
      {{259, 128, 196, 256}, {259, 128, 196, 256}},
      {{515, 256, 256, 512}, {515, 256, 384, 512}}};
  static const int FP_DIMS[4][3] = {{262, 128, 128}, {608, 256, 256}, {768, 512, 512}, {1536, 512, 512}};
  static const int NL[5] = {16384, 4096, 1024, 256, 64};
  static const int CL[5] = {6, 96, 256, 512, 1024};

  // ---- unpack params (setup_inputs() insertion order, DFS) ----
  const float* pc = (const float*)d_in[0];
  Layer sa[4][2][3];
  Layer fp[4][2];
  int q = 1;
  for (int i = 0; i < 4; ++i)
    for (int j = 0; j < 2; ++j)
      for (int l = 0; l < 3; ++l) {
        sa[i][j][l] = Layer{(const float*)d_in[q], (const float*)d_in[q + 1],
                            (const float*)d_in[q + 2], (const float*)d_in[q + 3]};
        q += 4;
      }
  for (int i = 0; i < 4; ++i)
    for (int l = 0; l < 2; ++l) {
      fp[i][l] = Layer{(const float*)d_in[q], (const float*)d_in[q + 1],
                       (const float*)d_in[q + 2], (const float*)d_in[q + 3]};
      q += 4;
    }
  const float* fc1W = (const float*)d_in[q++];
  const float* fc1b = (const float*)d_in[q++];
  const float* fc2W = (const float*)d_in[q++];
  const float* fc2b = (const float*)d_in[q++];

  // ---- workspace bump allocator ----
  char* wp = (char*)d_ws;
  auto alloc = [&](size_t bytes) -> void* {
    void* r = (void*)wp;
    wp += (bytes + 255) & ~(size_t)255;
    return r;
  };

  float* xyzL[5];
  float* featL[5];
  for (int i = 0; i < 5; ++i) {
    xyzL[i]  = (float*)alloc((size_t)Bn * NL[i] * 3 * 4);
    featL[i] = (float*)alloc((size_t)Bn * NL[i] * CL[i] * 4);
  }
  float* fpout[4];
  fpout[3] = (float*)alloc((size_t)Bn * NL[3] * 512 * 4);
  fpout[2] = (float*)alloc((size_t)Bn * NL[2] * 512 * 4);
  fpout[1] = (float*)alloc((size_t)Bn * NL[1] * 256 * 4);
  fpout[0] = (float*)d_out;  // gfeat [B,N,128]

  int*   fidx   = (int*)alloc((size_t)Bn * 4096 * 4);
  float* distws = (float*)alloc((size_t)Bn * Nn * 4);
  int*   bidx   = (int*)alloc((size_t)Bn * 4096 * 32 * 4);
  float* Ybuf   = (float*)alloc((size_t)Bn * 4096 * 32 * 64 * 4);          // 16.8M f32
  unsigned short* bfact = (unsigned short*)alloc((size_t)10 * 1024 * 1024 * 2);  // 10M bf16
  unsigned short* Wtbuf = (unsigned short*)alloc((size_t)1024 * 1024 * 2);       // 1M bf16
  float* meanb = (float*)alloc(512 * 4);
  float* rstdb = (float*)alloc(512 * 4);
  int*   knni  = (int*)alloc((size_t)Bn * Nn * 3 * 4);
  float* knnw  = (float*)alloc((size_t)Bn * Nn * 3 * 4);
  float* hbuf  = (float*)alloc((size_t)Bn * Nn * 128 * 4);

  auto run_gemm = [&](const unsigned short* Abf, const float* Wf, const float* bias,
                      float* Yout, int M, int Nc, int Kc, int relu) {
    const int Kpad = ceil32(Kc), Npad = ceil32(Nc);
    wt_convert_kernel<<<capped((long long)Npad * Kpad, 256, 4096), 256, 0, stream>>>(
        Wf, Kc, Nc, Kpad, Npad, Wtbuf);
    const long long tiles = (long long)(M >> 4) * (Npad >> 5);
    gemm_bf16_wmma<<<capped(tiles, 8, 16384), 256, 0, stream>>>(Abf, Wtbuf, bias, Yout, M, Nc,
                                                                Kpad, relu);
  };
  auto run_bn = [&](const Layer& L, float* dstF, unsigned short* dstB, int M, int Nc, int Kpad) {
    bn_stats_kernel<<<(unsigned)Nc, 256, 0, stream>>>(Ybuf, M, Nc, meanb, rstdb);
    bn_relu_dual_kernel<<<capped((long long)M * Kpad, 256, 65536), 256, 0, stream>>>(
        Ybuf, meanb, rstdb, L.g, L.beta, dstF, dstB, M, Nc, Kpad);
  };

  // ---- split input pointcloud ----
  split_pc_kernel<<<cdivu((long long)Bn * Nn, 256), 256, 0, stream>>>(pc, Bn * Nn, xyzL[0], featL[0]);

  // ---- SA modules ----
  for (int i = 0; i < 4; ++i) {
    const int np  = SA_NP[i];
    const int Nin = NL[i];
    const int Cin = CL[i];
    fps_kernel<<<Bn, 256, 0, stream>>>(xyzL[i], Nin, np, fidx, distws);
    gather_xyz_kernel<<<cdivu((long long)Bn * np, 256), 256, 0, stream>>>(xyzL[i], fidx, Bn * np,
                                                                          Nin, np, xyzL[i + 1]);
    int coff = 0;
    for (int j = 0; j < 2; ++j) {
      const int ns = SA_NS[j];
      const float r = SA_R[i][j];
      const long long centers = (long long)Bn * np;
      ball_query_kernel<<<cdivu(centers, 128), 128, 0, stream>>>(xyzL[i], xyzL[i + 1], Bn, Nin, np,
                                                                 ns, r * r, bidx);
      const long long rows = centers * ns;
      const int K0 = SA_DIMS[i][j][0];             // 3 + Cin
      const int K0pad = ceil32(K0);
      group_bf16_kernel<<<capped(rows * K0pad, 256, 65536), 256, 0, stream>>>(
          xyzL[i], featL[i], xyzL[i + 1], bidx, rows, Nin, np, ns, Cin, K0pad, bfact);

      const int M = (int)rows;
      for (int l = 0; l < 3; ++l) {
        const int Kc = SA_DIMS[i][j][l];
        const int Nc = SA_DIMS[i][j][l + 1];
        run_gemm(bfact, sa[i][j][l].W, sa[i][j][l].b, Ybuf, M, Nc, Kc, 0);
        if (l < 2) run_bn(sa[i][j][l], nullptr, bfact, M, Nc, ceil32(Nc));
        else       run_bn(sa[i][j][l], Ybuf, nullptr, M, Nc, Nc);
      }
      const int Cl = SA_DIMS[i][j][3];
      maxpool_kernel<<<capped(centers * Cl, 256, 65536), 256, 0, stream>>>(
          Ybuf, centers, ns, Cl, featL[i + 1], CL[i + 1], coff);
      coff += Cl;
    }
  }

  // ---- FP modules (deepest first) ----
  const float* knownFeat = featL[4];
  int knownC = CL[4];
  for (int s = 3; s >= 0; --s) {
    const int Nu = NL[s];
    const int Nk = NL[s + 1];
    const int Cs = CL[s];
    const int Ck = knownC;
    knn3_kernel<<<cdivu((long long)Bn * Nu, 128), 128, 0, stream>>>(xyzL[s], xyzL[s + 1], Bn, Nu,
                                                                    Nk, knni, knnw);
    const int K0pad = ceil32(Ck + Cs);
    interp_concat_bf16_kernel<<<capped((long long)Bn * Nu * K0pad, 256, 65536), 256, 0, stream>>>(
        knownFeat, knni, knnw, featL[s], Bn, Nu, Nk, Ck, Cs, K0pad, bfact);

    const int M = Bn * Nu;
    for (int l = 0; l < 2; ++l) {
      const int Kc = FP_DIMS[s][l];
      const int Nc = FP_DIMS[s][l + 1];
      run_gemm(bfact, fp[s][l].W, fp[s][l].b, Ybuf, M, Nc, Kc, 0);
      if (l == 0)      run_bn(fp[s][l], nullptr, bfact, M, Nc, ceil32(Nc));
      else if (s > 0)  run_bn(fp[s][l], fpout[s], nullptr, M, Nc, Nc);
      else             run_bn(fp[s][l], (float*)d_out, bfact, M, Nc, ceil32(Nc));  // gfeat + fc1 input
    }
    knownFeat = fpout[s];
    knownC = FP_DIMS[s][2];
  }

  // ---- head: fc1 (WMMA + fused relu) then fc2 dot ----
  {
    const int M = Bn * Nn;
    run_gemm(bfact, fc1W, fc1b, hbuf, M, 128, 128, 1);
    float* pred = (float*)d_out + (size_t)M * 128;
    fc2_kernel<<<cdivu((long long)M, 256), 256, 0, stream>>>(hbuf, fc2W, fc2b, pred, M);
  }
}
